// DialogEncoder_35734127903469
// MI455X (gfx1250) — matrix operand order repaired and checked by reading, not compiled, not run on hardware
//
#include <hip/hip_runtime.h>

// ---------------------------------------------------------------------------
// DialogEncoder for MI455X (gfx1250, wave32, WMMA, async global->LDS).
// L=4096, D1=1024, D2=512, NLAYERS=2.
// ---------------------------------------------------------------------------

#define LSEQ 4096
#define D1   1024
#define D2   512
#define D3   1536   // 3*D2 == D1+D2 (gru gates / concat width)

typedef __attribute__((ext_vector_type(16))) __bf16 v16bf;
typedef __attribute__((ext_vector_type(8)))  float  v8f;
typedef int v4i_vs __attribute__((vector_size(16)));   // pointee type of async-LDS builtin

union Frag16 { uint4 q[2]; v16bf v; };
union U4w    { uint4 q; unsigned short us[8]; };

#if defined(__has_builtin)
#if __has_builtin(__builtin_amdgcn_global_load_async_to_lds_b128)
#define HAVE_ASYNC_LDS 1
#endif
#if __has_builtin(__builtin_amdgcn_s_wait_asynccnt)
#define HAVE_WAIT_ASYNC 1
#endif
#endif

// 16-byte global -> LDS copy; async (ASYNCcnt-tracked, no VGPR round trip) when
// the gfx1250 builtin exists, else synchronous through registers.
__device__ __forceinline__ void copy16_g2l(const unsigned short* g, unsigned short* l) {
#if HAVE_ASYNC_LDS
    __builtin_amdgcn_global_load_async_to_lds_b128(
        (__attribute__((address_space(1))) v4i_vs*)(void*)g,
        (__attribute__((address_space(3))) v4i_vs*)(void*)l, 0, 0);
#else
    *(uint4*)l = *(const uint4*)g;
#endif
}
__device__ __forceinline__ void wait_async_zero() {
#if HAVE_ASYNC_LDS
#if HAVE_WAIT_ASYNC
    __builtin_amdgcn_s_wait_asynccnt(0);
#else
    asm volatile("s_wait_asynccnt 0" ::: "memory");
#endif
#endif
}

__device__ __forceinline__ unsigned short f2bf(float f) {
    unsigned u = __builtin_bit_cast(unsigned, f);
    unsigned r = u + 0x7FFFu + ((u >> 16) & 1u);   // round-to-nearest-even
    return (unsigned short)(r >> 16);
}
__device__ __forceinline__ float bf2f(unsigned short h) {
    return __builtin_bit_cast(float, (unsigned)h << 16);
}
__device__ __forceinline__ float sigmoidf(float x) { return 1.0f / (1.0f + expf(-x)); }

// ---------------------------------------------------------------------------
// f32 -> bf16 conversion (grid-stride)
// ---------------------------------------------------------------------------
__global__ void cvt_bf16(const float* __restrict__ in, unsigned short* __restrict__ out, int n) {
    for (int i = blockIdx.x * blockDim.x + threadIdx.x; i < n; i += gridDim.x * blockDim.x)
        out[i] = f2bf(in[i]);
}

// pack x (f32, 4096x1024) into concat buffer cols [512..1536) as bf16 (row stride D3)
__global__ void pack_x_concat(const float* __restrict__ x, unsigned short* __restrict__ cat) {
    for (int i = blockIdx.x * blockDim.x + threadIdx.x; i < LSEQ * D1; i += gridDim.x * blockDim.x) {
        int r = i >> 10, c = i & 1023;
        cat[(size_t)r * D3 + D2 + c] = f2bf(x[i]);
    }
}

// ---------------------------------------------------------------------------
// bf16 WMMA GEMM:  out[M,N] = act(A[M,K] @ B[N,K]^T + bias) (+ resid)
// 256 threads (8 waves), tile 128x64, K-chunk 32, double-buffered LDS with
// async global->LDS copies overlapping the WMMA burst of the previous chunk.
// A/B frags: contiguous K (k = 16*(lane/16) + s); LDS row stride 40 (80B) is conflict-free
// for B128 reads across 16 lanes. M mult of 128, N mult of 64, K mult of 32.
// ---------------------------------------------------------------------------
__global__ __launch_bounds__(256)
void gemm_bf16_wmma(const unsigned short* __restrict__ A,
                    const unsigned short* __restrict__ B,
                    const float* __restrict__ bias,
                    const float* __restrict__ resid,
                    float* __restrict__ outF,
                    unsigned short* __restrict__ outB,
                    int M, int N, int K, int act) {
    __shared__ unsigned short As[2][128 * 40];
    __shared__ unsigned short Bs[2][64 * 40];

    const int tid  = threadIdx.x;
    const int wave = tid >> 5;
    const int lane = tid & 31;
    const int m0 = blockIdx.y * 128;
    const int n0 = blockIdx.x * 64;

    // cooperative fill: 3 x B128 per thread per chunk (A: 512 segs, B: 256 segs)
    const int arow = tid >> 2;          // 0..63 (and +64 for second A segment)
    const int aseg = tid & 3;
    auto fill_tiles = [&](int buf, int k0) {
        copy16_g2l(&A[(size_t)(m0 + arow) * K + k0 + aseg * 8],
                   &As[buf][arow * 40 + aseg * 8]);
        copy16_g2l(&A[(size_t)(m0 + 64 + arow) * K + k0 + aseg * 8],
                   &As[buf][(64 + arow) * 40 + aseg * 8]);
        copy16_g2l(&B[(size_t)(n0 + arow) * K + k0 + aseg * 8],
                   &Bs[buf][arow * 40 + aseg * 8]);
    };

    v8f acc[4];
    const v8f z8 = {0.f,0.f,0.f,0.f,0.f,0.f,0.f,0.f};
    acc[0] = z8; acc[1] = z8; acc[2] = z8; acc[3] = z8;

    const int lr    = lane & 15;
    const int koff  = (lane >> 4) << 4;   // A frag: K base 0/16
    const int kbase = (lane >> 4) << 4;   // B frag: K base 0/16

    const int chunks = K >> 5;
    fill_tiles(0, 0);
    wait_async_zero();
    __syncthreads();

    for (int c = 0; c < chunks; ++c) {
        const int buf = c & 1;
        if (c + 1 < chunks) fill_tiles(buf ^ 1, (c + 1) << 5);

        // preload all fragments, then a clean 4-WMMA burst (indep. accumulators)
        Frag16 af;
        af.q[0] = *(const uint4*)&As[buf][(wave * 16 + lr) * 40 + koff];
        af.q[1] = *(const uint4*)&As[buf][(wave * 16 + lr) * 40 + koff + 8];
        Frag16 bf0, bf1, bf2, bf3;
        bf0.q[0] = *(const uint4*)&Bs[buf][(lr) * 40 + kbase];
        bf0.q[1] = *(const uint4*)&Bs[buf][(lr) * 40 + kbase + 8];
        bf1.q[0] = *(const uint4*)&Bs[buf][(16 + lr) * 40 + kbase];
        bf1.q[1] = *(const uint4*)&Bs[buf][(16 + lr) * 40 + kbase + 8];
        bf2.q[0] = *(const uint4*)&Bs[buf][(32 + lr) * 40 + kbase];
        bf2.q[1] = *(const uint4*)&Bs[buf][(32 + lr) * 40 + kbase + 8];
        bf3.q[0] = *(const uint4*)&Bs[buf][(48 + lr) * 40 + kbase];
        bf3.q[1] = *(const uint4*)&Bs[buf][(48 + lr) * 40 + kbase + 8];

        acc[0] = __builtin_amdgcn_wmma_f32_16x16x32_bf16(false, af.v, false, bf0.v, (short)0, acc[0], false, false);
        acc[1] = __builtin_amdgcn_wmma_f32_16x16x32_bf16(false, af.v, false, bf1.v, (short)0, acc[1], false, false);
        acc[2] = __builtin_amdgcn_wmma_f32_16x16x32_bf16(false, af.v, false, bf2.v, (short)0, acc[2], false, false);
        acc[3] = __builtin_amdgcn_wmma_f32_16x16x32_bf16(false, af.v, false, bf3.v, (short)0, acc[3], false, false);

        wait_async_zero();   // next-chunk tiles resident before the barrier
        __syncthreads();
    }

    // C layout: VGPR i -> M = i + (lane>=16 ? 8 : 0); N = lane&15 per 16-col block
    const int rbase = m0 + wave * 16 + ((lane >> 4) << 3);
    #pragma unroll
    for (int nb = 0; nb < 4; ++nb) {
        int col = n0 + nb * 16 + (lane & 15);
        float bv = bias ? bias[col] : 0.f;
        #pragma unroll
        for (int i = 0; i < 8; ++i) {
            int row = rbase + i;
            float v = acc[nb][i] + bv;
            if (act == 1) v = tanhf(v);
            if (resid) v += resid[(size_t)row * N + col];
            if (outF)  outF[(size_t)row * N + col] = v;
            if (outB)  outB[(size_t)row * N + col] = f2bf(v);
        }
    }
}

// ---------------------------------------------------------------------------
// Sequential GRU scan: one workgroup (1024 thr). h, gh in LDS; Whh bf16 (L2-hot).
// gi: [L, 1536] f32 (precomputed x@Wih^T+bih). Writes h_t into concat cols [0,512) bf16.
// ---------------------------------------------------------------------------
__global__ __launch_bounds__(1024)
void gru_scan(const float* __restrict__ gi,
              const unsigned short* __restrict__ Whh,   // [1536,512] bf16
              const float* __restrict__ bhh,            // [1536]
              unsigned short* __restrict__ cat) {       // [L, D3] bf16
    __shared__ float hsm[D2];
    __shared__ float ghs[D3];
    const int tid = threadIdx.x;
    if (tid < D2) hsm[tid] = 0.f;
    __syncthreads();

    for (int t = 0; t < LSEQ; ++t) {
        // gh = h @ Whh^T + bhh  (1536 rows over 1024 threads)
        for (int row = tid; row < D3; row += 1024) {
            const unsigned short* w = Whh + (size_t)row * D2;
            float acc = 0.f;
            for (int k = 0; k < D2; k += 8) {
                U4w u; u.q = *(const uint4*)&w[k];
                acc += hsm[k+0]*bf2f(u.us[0]) + hsm[k+1]*bf2f(u.us[1])
                     + hsm[k+2]*bf2f(u.us[2]) + hsm[k+3]*bf2f(u.us[3])
                     + hsm[k+4]*bf2f(u.us[4]) + hsm[k+5]*bf2f(u.us[5])
                     + hsm[k+6]*bf2f(u.us[6]) + hsm[k+7]*bf2f(u.us[7]);
            }
            ghs[row] = acc + bhh[row];
        }
        __syncthreads();
        if (tid < D2) {
            const float* g = gi + (size_t)t * D3;
            float ir = g[tid], iz = g[D2 + tid], in_ = g[2*D2 + tid];
            float r = sigmoidf(ir + ghs[tid]);
            float z = sigmoidf(iz + ghs[D2 + tid]);
            float n = tanhf(in_ + r * ghs[2*D2 + tid]);
            float hn = (1.f - z) * n + z * hsm[tid];
            hsm[tid] = hn;
            cat[(size_t)t * D3 + tid] = f2bf(hn);
        }
        __syncthreads();
    }
}

// ---------------------------------------------------------------------------
// u = W^T @ wc  (column dots), u[1024] = dot(b, wc)   -- collapses sa/sb GEMMs
// grid 4 x 256
// ---------------------------------------------------------------------------
__global__ __launch_bounds__(256)
void build_u(const float* __restrict__ W, const float* __restrict__ wc,
             const float* __restrict__ b, float* __restrict__ u) {
    int e = blockIdx.x * 256 + threadIdx.x;
    float acc = 0.f;
    for (int d = 0; d < D1; ++d) acc += W[(size_t)d * D1 + e] * wc[d];
    u[e] = acc;
    if (blockIdx.x == 0) {
        __shared__ float red[256];
        float pb = 0.f;
        for (int d = threadIdx.x; d < D1; d += 256) pb += b[d] * wc[d];
        red[threadIdx.x] = pb;
        __syncthreads();
        for (int s = 128; s > 0; s >>= 1) {
            if (threadIdx.x < s) red[threadIdx.x] += red[threadIdx.x + s];
            __syncthreads();
        }
        if (threadIdx.x == 0) u[D1] = red[0];
    }
}

// sa[row] = H[row]·ua + ua[1024];  sb[row] = H[row]·ub + ub[1024]; one wave/row
__global__ __launch_bounds__(256)
void row_dots(const float* __restrict__ H, const float* __restrict__ ua,
              const float* __restrict__ ub, float* __restrict__ sa, float* __restrict__ sb) {
    int wave = threadIdx.x >> 5, lane = threadIdx.x & 31;
    int row = blockIdx.x * 8 + wave;
    const float* h = H + (size_t)row * D1;
    float a = 0.f, b = 0.f;
    for (int d = lane; d < D1; d += 32) { float hv = h[d]; a += hv * ua[d]; b += hv * ub[d]; }
    for (int off = 16; off > 0; off >>= 1) { a += __shfl_down(a, off); b += __shfl_down(b, off); }
    if (lane == 0) { sa[row] = a + ua[D1]; sb[row] = b + ub[D1]; }
}

// per-row softmax stats over j of leaky(sa_i + sb_j + bc): rowMax, rowExpSum
__global__ __launch_bounds__(256)
void softmax_stats(const float* __restrict__ sa, const float* __restrict__ sb,
                   const float* __restrict__ bcp, float* __restrict__ mOut,
                   float* __restrict__ sOut) {
    __shared__ float sbs[LSEQ];
    for (int i = threadIdx.x; i < LSEQ; i += 256) sbs[i] = sb[i];
    __syncthreads();
    int wave = threadIdx.x >> 5, lane = threadIdx.x & 31;
    int row = blockIdx.x * 8 + wave;
    float sav = sa[row], bc = bcp[0];
    float mx = -3.4e38f;
    for (int j = lane; j < LSEQ; j += 32) {
        float l = sav + sbs[j] + bc; l = (l >= 0.f) ? l : 0.01f * l;
        mx = fmaxf(mx, l);
    }
    for (int off = 16; off > 0; off >>= 1) mx = fmaxf(mx, __shfl_down(mx, off));
    mx = __shfl(mx, 0);
    float sum = 0.f;
    for (int j = lane; j < LSEQ; j += 32) {
        float l = sav + sbs[j] + bc; l = (l >= 0.f) ? l : 0.01f * l;
        sum += expf(l - mx);
    }
    for (int off = 16; off > 0; off >>= 1) sum += __shfl_down(sum, off);
    if (lane == 0) { mOut[row] = mx; sOut[row] = sum; }
}

// ---------------------------------------------------------------------------
// Flash-style attention apply: out[i,d] = sum_j P_ij * V[j,d], P built on the fly
// from rank-1 logits (never materializes 4096x4096 c). WMMA bf16.
// block 256 (8 waves): i-tile 16, d-tile 128 (16 cols/wave). grid (8, 256).
// ---------------------------------------------------------------------------
__global__ __launch_bounds__(256)
void attn_apply(const float* __restrict__ sa, const float* __restrict__ sb,
                const float* __restrict__ bcp, const float* __restrict__ mrow,
                const float* __restrict__ srow, const unsigned short* __restrict__ V,
                unsigned short* __restrict__ outB) {
    __shared__ float sa_s[16], m_s[16], inv_s[16];
    __shared__ float sb_s[LSEQ];
    __shared__ unsigned short p_s[16 * 40];
    __shared__ unsigned short vT[128 * 40];   // [d_local][k], k-contiguous for B frags

    const int tid  = threadIdx.x;
    const int wave = tid >> 5, lane = tid & 31;
    const int d0 = blockIdx.x * 128;
    const int i0 = blockIdx.y * 16;
    const float bc = bcp[0];

    if (tid < 16) {
        sa_s[tid]  = sa[i0 + tid];
        m_s[tid]   = mrow[i0 + tid];
        inv_s[tid] = 1.0f / srow[i0 + tid];
    }
    for (int i = tid; i < LSEQ; i += 256) sb_s[i] = sb[i];
    __syncthreads();

    v8f acc;
    { const v8f z8 = {0.f,0.f,0.f,0.f,0.f,0.f,0.f,0.f}; acc = z8; }

    const int lr    = lane & 15;
    const int koff  = (lane >> 4) << 4;
    const int kbase = (lane >> 4) << 4;

    for (int j0 = 0; j0 < LSEQ; j0 += 32) {
        // build normalized P tile (16 x 32) in LDS
        #pragma unroll
        for (int half = 0; half < 2; ++half) {
            int idx = tid + half * 256;
            int r = idx >> 5, jj = idx & 31;
            float l = sa_s[r] + sb_s[j0 + jj] + bc;
            l = (l >= 0.f) ? l : 0.01f * l;
            p_s[r * 40 + jj] = f2bf(expf(l - m_s[r]) * inv_s[r]);
        }
        // stage V tile transposed: vT[d][k] = V[j0+k][d0+d]
        #pragma unroll
        for (int half = 0; half < 2; ++half) {
            int idx = tid + half * 256;
            int k = idx >> 4, seg = idx & 15;
            U4w u; u.q = *(const uint4*)&V[(size_t)(j0 + k) * D1 + d0 + seg * 8];
            #pragma unroll
            for (int e = 0; e < 8; ++e) vT[(seg * 8 + e) * 40 + k] = u.us[e];
        }
        __syncthreads();

        Frag16 af;
        af.q[0] = *(const uint4*)&p_s[lr * 40 + koff];
        af.q[1] = *(const uint4*)&p_s[lr * 40 + koff + 8];
        Frag16 bfr;
        bfr.q[0] = *(const uint4*)&vT[(wave * 16 + lr) * 40 + kbase];
        bfr.q[1] = *(const uint4*)&vT[(wave * 16 + lr) * 40 + kbase + 8];
        acc = __builtin_amdgcn_wmma_f32_16x16x32_bf16(
            false, af.v, false, bfr.v, (short)0, acc, false, false);
        __syncthreads();
    }

    const int rbase = i0 + ((lane >> 4) << 3);
    const int col   = d0 + wave * 16 + (lane & 15);
    #pragma unroll
    for (int i = 0; i < 8; ++i)
        outB[(size_t)(rbase + i) * D1 + col] = f2bf(acc[i]);
}

// ---------------------------------------------------------------------------
// Host-side orchestration
// ---------------------------------------------------------------------------
extern "C" void kernel_launch(void* const* d_in, const int* in_sizes, int n_in,
                              void* d_out, int out_size, void* d_ws, size_t ws_size,
                              hipStream_t stream) {
    (void)in_sizes; (void)n_in; (void)out_size; (void)ws_size;

    const float* x       = (const float*)d_in[0];
    const float* gru_Wih = (const float*)d_in[1];
    const float* gru_Whh = (const float*)d_in[2];
    const float* gru_bih = (const float*)d_in[3];
    const float* gru_bhh = (const float*)d_in[4];
    const float* fc_W    = (const float*)d_in[5];
    const float* fc_b    = (const float*)d_in[6];
    const float* gat_Wa  = (const float*)d_in[7];
    const float* gat_ba  = (const float*)d_in[8];
    const float* gat_Wb  = (const float*)d_in[9];
    const float* gat_bb  = (const float*)d_in[10];
    const float* gat_Wv  = (const float*)d_in[11];
    const float* gat_bv  = (const float*)d_in[12];
    const float* gat_Wc  = (const float*)d_in[13];
    const float* gat_bc  = (const float*)d_in[14];
    const float* gat_Wm  = (const float*)d_in[15];
    const float* gat_bm  = (const float*)d_in[16];
    const float* out_W   = (const float*)d_in[17];
    const float* out_b   = (const float*)d_in[18];
    float* out = (float*)d_out;

    char* ws = (char*)d_ws;
    size_t off = 0;
    auto take = [&](size_t bytes) -> char* {
        char* p = ws + off;
        off += (bytes + 255) & ~(size_t)255;
        return p;
    };

    unsigned short* Wih_bf  = (unsigned short*)take((size_t)D3 * D1 * 2);
    unsigned short* Whh_bf  = (unsigned short*)take((size_t)D3 * D2 * 2);
    unsigned short* fcW_bf  = (unsigned short*)take((size_t)D1 * D3 * 2);
    unsigned short* Wv_bf   = (unsigned short*)take((size_t)2 * D1 * D1 * 2);
    unsigned short* Wm_bf   = (unsigned short*)take((size_t)2 * D1 * D1 * 2);
    unsigned short* outW_bf = (unsigned short*)take((size_t)D1 * D1 * 2);
    unsigned short* x_bf    = (unsigned short*)take((size_t)LSEQ * D1 * 2);
    float*          gi      = (float*)take((size_t)LSEQ * D3 * 4);
    unsigned short* cat     = (unsigned short*)take((size_t)LSEQ * D3 * 2);
    float*          curF    = (float*)take((size_t)LSEQ * D1 * 4);
    unsigned short* curB    = (unsigned short*)take((size_t)LSEQ * D1 * 2);
    unsigned short* vB      = (unsigned short*)take((size_t)LSEQ * D1 * 2);
    unsigned short* attnB   = (unsigned short*)take((size_t)LSEQ * D1 * 2);
    float*          ua      = (float*)take((D1 + 1) * 4);
    float*          ub      = (float*)take((D1 + 1) * 4);
    float*          sa      = (float*)take(LSEQ * 4);
    float*          sb      = (float*)take(LSEQ * 4);
    float*          mrow    = (float*)take(LSEQ * 4);
    float*          srow    = (float*)take(LSEQ * 4);

    auto cvt = [&](const float* in, unsigned short* o, size_t n) {
        int blocks = (int)((n + 2047) / 2048);
        cvt_bf16<<<blocks, 256, 0, stream>>>(in, o, (int)n);
    };

    // 1) one-time per-launch bf16 conversions
    cvt(gru_Wih, Wih_bf, (size_t)D3 * D1);
    cvt(gru_Whh, Whh_bf, (size_t)D3 * D2);
    cvt(fc_W,    fcW_bf, (size_t)D1 * D3);
    cvt(gat_Wv,  Wv_bf,  (size_t)2 * D1 * D1);
    cvt(gat_Wm,  Wm_bf,  (size_t)2 * D1 * D1);
    cvt(out_W,   outW_bf,(size_t)D1 * D1);
    cvt(x,       x_bf,   (size_t)LSEQ * D1);
    pack_x_concat<<<2048, 256, 0, stream>>>(x, cat);

    // 2) gi = x @ Wih^T + bih         [4096,1536]
    gemm_bf16_wmma<<<dim3(D3 / 64, LSEQ / 128), 256, 0, stream>>>(
        x_bf, Wih_bf, gru_bih, nullptr, gi, nullptr, LSEQ, D3, D1, 0);

    // 3) sequential GRU scan -> concat cols [0,512)
    gru_scan<<<1, 1024, 0, stream>>>(gi, Whh_bf, gru_bhh, cat);

    // 4) cur = tanh(concat @ fc_W^T + fc_b)
    gemm_bf16_wmma<<<dim3(D1 / 64, LSEQ / 128), 256, 0, stream>>>(
        cat, fcW_bf, fc_b, nullptr, curF, curB, LSEQ, D1, D3, 1);

    // 5) GAT layers
    for (int l = 0; l < 2; ++l) {
        const float* Wa_l = gat_Wa + (size_t)l * D1 * D1;
        const float* Wb_l = gat_Wb + (size_t)l * D1 * D1;
        const float* ba_l = gat_ba + (size_t)l * D1;
        const float* bb_l = gat_bb + (size_t)l * D1;
        const float* bv_l = gat_bv + (size_t)l * D1;
        const float* bm_l = gat_bm + (size_t)l * D1;
        const float* wc_a = gat_Wc + (size_t)l * 2 * D1;
        const float* wc_b = wc_a + D1;
        const float* bc_l = gat_bc + l;
        const unsigned short* Wv_l = Wv_bf + (size_t)l * D1 * D1;
        const unsigned short* Wm_l = Wm_bf + (size_t)l * D1 * D1;

        // sa/sb collapse: u = W^T wc (+ bias dot), then matvec
        build_u<<<4, 256, 0, stream>>>(Wa_l, wc_a, ba_l, ua);
        build_u<<<4, 256, 0, stream>>>(Wb_l, wc_b, bb_l, ub);
        row_dots<<<LSEQ / 8, 256, 0, stream>>>(curF, ua, ub, sa, sb);

        // v = cur @ Wv^T + bv (bf16 for attention B operand)
        gemm_bf16_wmma<<<dim3(D1 / 64, LSEQ / 128), 256, 0, stream>>>(
            curB, Wv_l, bv_l, nullptr, nullptr, vB, LSEQ, D1, D1, 0);

        // softmax stats + flash-style P@V (c never materialized)
        softmax_stats<<<LSEQ / 8, 256, 0, stream>>>(sa, sb, bc_l, mrow, srow);
        attn_apply<<<dim3(D1 / 128, LSEQ / 16), 256, 0, stream>>>(
            sa, sb, bc_l, mrow, srow, vB, attnB);

        // cur = tanh(attn @ Wm^T + bm) + cur   (residual fused)
        gemm_bf16_wmma<<<dim3(D1 / 64, LSEQ / 128), 256, 0, stream>>>(
            attnB, Wm_l, bm_l, curF, curF, curB, LSEQ, D1, D1, 1);
    }

    // 6) out = tanh(cur @ out_W^T + out_b)
    gemm_bf16_wmma<<<dim3(D1 / 64, LSEQ / 128), 256, 0, stream>>>(
        curB, outW_bf, out_b, nullptr, out, nullptr, LSEQ, D1, D1, 1);
}
